// Mamba2Simple_82910048681998
// MI455X (gfx1250) — compile-verified, matrix operand order checked by reading, not run on hardware
//
#include <hip/hip_runtime.h>
#include <hip/hip_bf16.h>

// ---------------- problem constants ----------------
#define D_MODEL   1024
#define D_STATE   128
#define D_CONV    4
#define HEADDIM   128
#define CHUNK     256
#define BATCH     2
#define SEQLEN    2048
#define D_INNER   2048              // EXPAND * D_MODEL
#define NHEADS    16                // D_INNER / HEADDIM
#define D_IN_PROJ 4368              // 2*D_INNER + 2*D_STATE + NHEADS
#define CONV_DIM  2304              // D_INNER + 2*D_STATE
#define NCHUNK    8                 // SEQLEN / CHUNK
#define MROWS     (BATCH*SEQLEN)    // 4096
#define EPS       1e-5f

typedef __bf16 bf16;
typedef __attribute__((ext_vector_type(16))) __bf16 v16bf;
typedef __attribute__((ext_vector_type(8)))  __bf16 v8bf;
typedef __attribute__((ext_vector_type(4)))  __bf16 v4bf;
typedef __attribute__((ext_vector_type(8)))  float  v8f;

// ---------------------------------------------------------------------
// WMMA fragment loader (ISA 7.12.2). Tile stored row-major [16][ld]
// where "row" is the M dim (A operand) or N dim (B operand), ld = K
// stride in elements (must be multiple of 8 -> 16B aligned rows).
// Lane (r, half) needs cols {8h..8h+7} and {16+8h..23+8h}: two b128 loads.
// ---------------------------------------------------------------------
__device__ __forceinline__ v16bf load_frag(const bf16* p, int ld) {
  const int lane = threadIdx.x & 31;
  const int row = lane & 15, half = lane >> 4;
  const bf16* q = p + row * ld + (half << 3);
  v8bf lo = *(const v8bf*)q;          // ds_load_b128
  v8bf hi = *(const v8bf*)(q + 16);   // ds_load_b128
  return __builtin_shufflevector(lo, hi, 0, 1, 2, 3, 4, 5, 6, 7, 8, 9, 10, 11,
                                 12, 13, 14, 15);
}

__device__ __forceinline__ v8f wmma_bf16(v16bf a, v16bf b, v8f c) {
  return __builtin_amdgcn_wmma_f32_16x16x32_bf16(false, a, false, b, (short)0, c,
                                                 false, false);
}

__device__ __forceinline__ void st_bf16x4(bf16* dst, float4 v) {
  v4bf t;
  t[0] = (bf16)v.x; t[1] = (bf16)v.y; t[2] = (bf16)v.z; t[3] = (bf16)v.w;
  *(v4bf*)dst = t;                    // ds_store_b64
}

__device__ __forceinline__ float siluf(float x) { return x / (1.f + __expf(-x)); }
__device__ __forceinline__ float softplusf(float x) {
  return (x > 20.f) ? x : log1pf(__expf(x));
}

// =====================================================================
// K1/K7: C[M,N] = A[M,K] * W[N,K]^T   (f32 in, bf16 WMMA, f32 out)
// block 256 = 8 waves; tile 64x64, BK=64; both LDS tiles in [row][K] form
// =====================================================================
#define BM 64
#define BN 64
#define BK 64
__global__ __launch_bounds__(256, 2)
void gemm_nt_kernel(const float* __restrict__ A, const float* __restrict__ W,
                    float* __restrict__ C, int M, int N, int K) {
  __shared__ bf16 Al[BM * BK];   // [m][k]
  __shared__ bf16 Bl[BN * BK];   // [n][k]  (transposed stage -> coalesced)
  const int tid  = threadIdx.x;
  const int wave = tid >> 5;
  const int lane = tid & 31;
  const int m0 = blockIdx.y * BM;
  const int n0 = blockIdx.x * BN;
  const int mi  = wave & 3;          // m-tile 0..3
  const int njb = (wave >> 2) * 2;   // n-tiles njb, njb+1

  const int sr = tid >> 2;           // 0..63 staging row
  const int sc = (tid & 3) * 16;     // staging col base (16 floats)

  v8f acc0 = {}, acc1 = {};
  for (int k0 = 0; k0 < K; k0 += BK) {
    // stage A (64x64): 16 consecutive floats per thread, 4x float4
    {
      const float* src = &A[(size_t)(m0 + sr) * K + k0 + sc];
#pragma unroll
      for (int j = 0; j < 4; ++j)
        st_bf16x4(&Al[sr * BK + sc + 4 * j], *(const float4*)(src + 4 * j));
    }
    // stage B transposed: Bl[n][k] = W[n0+n][k0+k] (coalesced along K)
    if (n0 + sr < N) {
      const float* src = &W[(size_t)(n0 + sr) * K + k0 + sc];
#pragma unroll
      for (int j = 0; j < 4; ++j)
        st_bf16x4(&Bl[sr * BK + sc + 4 * j], *(const float4*)(src + 4 * j));
    } else {
      v4bf z = {};
#pragma unroll
      for (int j = 0; j < 4; ++j) *(v4bf*)&Bl[sr * BK + sc + 4 * j] = z;
    }
    if (k0 + BK < K)
      __builtin_prefetch(&A[(size_t)(m0 + sr) * K + k0 + BK + sc], 0, 0);
    __syncthreads();

    v16bf af0 = load_frag(Al + mi * 16 * BK, BK);
    v16bf af1 = load_frag(Al + mi * 16 * BK + 32, BK);
    v16bf b00 = load_frag(Bl + (njb * 16) * BK, BK);
    v16bf b01 = load_frag(Bl + (njb * 16) * BK + 32, BK);
    v16bf b10 = load_frag(Bl + ((njb + 1) * 16) * BK, BK);
    v16bf b11 = load_frag(Bl + ((njb + 1) * 16) * BK + 32, BK);
    acc0 = wmma_bf16(af0, b00, acc0);
    acc0 = wmma_bf16(af1, b01, acc0);
    acc1 = wmma_bf16(af0, b10, acc1);
    acc1 = wmma_bf16(af1, b11, acc1);
    __syncthreads();
  }
  const int half = lane >> 4, nc = lane & 15;
#pragma unroll
  for (int r = 0; r < 8; ++r) {
    const int m  = m0 + mi * 16 + r + 8 * half;
    const int na = n0 + njb * 16 + nc;
    const int nb = n0 + (njb + 1) * 16 + nc;
    if (na < N) C[(size_t)m * N + na] = acc0[r];
    if (nb < N) C[(size_t)m * N + nb] = acc1[r];
  }
}

// =====================================================================
// K2: depthwise causal conv1d (width 4) + bias + SiLU
// =====================================================================
__global__ __launch_bounds__(256)
void conv_silu_kernel(const float* __restrict__ zxbcdt,
                      const float* __restrict__ conv_w,
                      const float* __restrict__ conv_b,
                      float* __restrict__ xBC_act) {
  const int idx = blockIdx.x * blockDim.x + threadIdx.x;
  const int total = BATCH * SEQLEN * CONV_DIM;
  if (idx >= total) return;
  const int ch = idx % CONV_DIM;
  const int l  = (idx / CONV_DIM) % SEQLEN;
  const int bb = idx / (CONV_DIM * SEQLEN);
  float acc = conv_b[ch];
#pragma unroll
  for (int j = 0; j < D_CONV; ++j) {
    const int ll = l - (D_CONV - 1) + j;
    if (ll >= 0) {
      const float xv =
          zxbcdt[((size_t)(bb * SEQLEN + ll)) * D_IN_PROJ + D_INNER + ch];
      acc += conv_w[ch * D_CONV + j] * xv;
    }
  }
  xBC_act[idx] = siluf(acc);
}

// =====================================================================
// K3: per (chunk, head, batch) SSD intra-chunk:  Y_diag + chunk states
// block = 512 threads = 16 waves, dynamic LDS ~287KB (one WG per WGP)
// Tiles:
//   Ctl [t=256][d], Btl [s=256][d]         (stride LDC=136)
//   XdtT [p=128][s], BtlT=dcy*B [d=128][s] (stride LDT=264)
// so every WMMA operand is the contiguous two-b128 pattern.
// =====================================================================
#define LDC 136
#define LDT 264
__global__ __launch_bounds__(512, 1)
void ssd_chunk_kernel(const float* __restrict__ zxbcdt,
                      const float* __restrict__ xBC_act,
                      const float* __restrict__ dt_bias,
                      const float* __restrict__ A_log,
                      float* __restrict__ Ydiag,
                      float* __restrict__ states,
                      float* __restrict__ AcsG) {
  extern __shared__ char smem[];
  float* dtv = (float*)smem;             // [256]
  float* acs = dtv + CHUNK;              // [256]
  float* dcy = acs + CHUNK;              // [256]
  bf16* Ctl  = (bf16*)(dcy + CHUNK);     // [256][LDC]
  bf16* Btl  = Ctl + CHUNK * LDC;        // [256][LDC]
  bf16* XdtT = Btl + CHUNK * LDC;        // [128][LDT]
  bf16* BtlT = XdtT + 128 * LDT;         // [128][LDT]
  bf16* ssc  = BtlT + 128 * LDT;         // [16 waves][16][32]

  const int c = blockIdx.x, h = blockIdx.y, b = blockIdx.z;
  const int tid = threadIdx.x, wave = tid >> 5, lane = tid & 31;
  const int half = lane >> 4, nn = lane & 15;
  const size_t row0 = (size_t)b * SEQLEN + (size_t)c * CHUNK;

  // ---- dt softplus + dA, then Hillis-Steele inclusive cumsum ----
  const float Aneg = -__expf(A_log[h]);
  if (tid < CHUNK) {
    const float raw = zxbcdt[(row0 + tid) * D_IN_PROJ + (D_IN_PROJ - NHEADS) + h];
    const float d   = softplusf(raw + dt_bias[h]);
    dtv[tid] = d;
    acs[tid] = d * Aneg;
  }
  __syncthreads();
  for (int off = 1; off < CHUNK; off <<= 1) {
    float v = 0.f, a = 0.f;
    if (tid < CHUNK) { v = acs[tid]; if (tid >= off) a = acs[tid - off]; }
    __syncthreads();
    if (tid < CHUNK) acs[tid] = v + a;
    __syncthreads();
  }
  if (tid < CHUNK) {
    dcy[tid] = __expf(acs[CHUNK - 1] - acs[tid]);
    AcsG[(((size_t)(b * NCHUNK + c) * NHEADS + h) << 8) + tid] = acs[tid];
  }
  __syncthreads();

  // ---- stage Ctl/Btl ([row][d], float4 -> 4x bf16) ----
  for (int i = tid; i < CHUNK * 32; i += 512) {
    const int s = i >> 5, q = (i & 31) * 4;
    const size_t rb = (row0 + s) * CONV_DIM + D_INNER;
    st_bf16x4(&Btl[s * LDC + q], *(const float4*)&xBC_act[rb + q]);
    st_bf16x4(&Ctl[s * LDC + q], *(const float4*)&xBC_act[rb + D_STATE + q]);
  }
  // ---- stage XdtT[p][s] = x*dt (transposed), BtlT[d][s] = dcy*B ----
  for (int i = tid; i < 128 * 64; i += 512) {
    const int p = i >> 6, sq = (i & 63) * 4;
    float4 xv, bv;
    float* xf = (float*)&xv;
    float* bf = (float*)&bv;
#pragma unroll
    for (int j = 0; j < 4; ++j) {
      const size_t rb = (row0 + sq + j) * CONV_DIM;
      xf[j] = xBC_act[rb + h * HEADDIM + p] * dtv[sq + j];
      bf[j] = xBC_act[rb + D_INNER + p] * dcy[sq + j];
    }
    st_bf16x4(&XdtT[p * LDT + sq], xv);
    st_bf16x4(&BtlT[p * LDT + sq], bv);
  }
  __syncthreads();

  // ---- Y_diag: wave m owns query rows [16m,16m+16) ----
  const int m = wave;
  v8f accY[8] = {};
  bf16* myss = ssc + wave * (16 * 32);
  const int spmax = ((m << 4) + 15) >> 5;
  for (int sp = 0; sp <= spmax; ++sp) {
    v8f s0 = {}, s1 = {};
#pragma unroll
    for (int kk = 0; kk < D_STATE; kk += 32) {
      v16bf cf = load_frag(Ctl + (m * 16) * LDC + kk, LDC);
      s0 = wmma_bf16(cf, load_frag(Btl + (sp * 32) * LDC + kk, LDC), s0);
      s1 = wmma_bf16(cf, load_frag(Btl + (sp * 32 + 16) * LDC + kk, LDC), s1);
    }
    // causal mask + segment decay; reroute D-layout -> A-layout via LDS
    const int sA = sp * 32 + nn, sB = sp * 32 + 16 + nn;
#pragma unroll
    for (int r = 0; r < 8; ++r) {
      const int tt = m * 16 + r + 8 * half;
      const float ea = (sA <= tt) ? s0[r] * __expf(acs[tt] - acs[sA]) : 0.f;
      const float eb = (sB <= tt) ? s1[r] * __expf(acs[tt] - acs[sB]) : 0.f;
      myss[(r + 8 * half) * 32 + nn]      = (bf16)ea;
      myss[(r + 8 * half) * 32 + 16 + nn] = (bf16)eb;
    }
    v16bf sf = load_frag(myss, 32);
#pragma unroll
    for (int pj = 0; pj < 8; ++pj)
      accY[pj] = wmma_bf16(
          sf, load_frag(XdtT + (pj * 16) * LDT + sp * 32, LDT), accY[pj]);
  }
#pragma unroll
  for (int pj = 0; pj < 8; ++pj)
#pragma unroll
    for (int r = 0; r < 8; ++r) {
      const int tt = m * 16 + r + 8 * half;
      Ydiag[(row0 + tt) * D_INNER + h * HEADDIM + pj * 16 + nn] = accY[pj][r];
    }

  // ---- chunk states[p,n] = sum_s (dcy*B)[s,n] * Xdt[s,p] ----
  const int pi = wave >> 1;
  const int nb = (wave & 1) * 4;  // 4 n-tiles per wave
  v8f st[4] = {};
  for (int kk = 0; kk < CHUNK; kk += 32) {
    v16bf af = load_frag(XdtT + (pi * 16) * LDT + kk, LDT);
#pragma unroll
    for (int j = 0; j < 4; ++j)
      st[j] = wmma_bf16(af, load_frag(BtlT + ((nb + j) * 16) * LDT + kk, LDT),
                        st[j]);
  }
  const size_t sb = ((size_t)(b * NCHUNK + c) * NHEADS + h) << 14;  // *128*128
#pragma unroll
  for (int j = 0; j < 4; ++j)
#pragma unroll
    for (int r = 0; r < 8; ++r)
      states[sb + (size_t)(pi * 16 + r + 8 * half) * 128 + (nb + j) * 16 + nn] =
          st[j][r];
}

// =====================================================================
// K4: sequential inter-chunk state scan (per batch,head)
// =====================================================================
__global__ __launch_bounds__(256)
void chunk_scan_kernel(const float* __restrict__ states,
                       const float* __restrict__ AcsG,
                       float* __restrict__ prev) {
  const int bh = blockIdx.x;          // 0..31
  const int bb = bh >> 4, h = bh & 15;
  for (int idx = threadIdx.x; idx < 128 * 128; idx += 256) {
    float run = 0.f;
    for (int cc = 0; cc < NCHUNK; ++cc) {
      const size_t base = (((size_t)(bb * NCHUNK + cc) * NHEADS + h) << 14) + idx;
      prev[base] = run;
      const float dA =
          AcsG[(((size_t)(bb * NCHUNK + cc) * NHEADS + h) << 8) + (CHUNK - 1)];
      run = __expf(dA) * run + states[base];
    }
  }
}

// =====================================================================
// K5: Y_off = decay(t) * C . prev^T  (WMMA) + D-skip + SiLU(z) gate
// Pt staged [p][n] == natural layout of prev -> contiguous b128 frags
// =====================================================================
__global__ __launch_bounds__(512, 1)
void ssd_off_gate_kernel(const float* __restrict__ zxbcdt,
                         const float* __restrict__ xBC_act,
                         const float* __restrict__ prev,
                         const float* __restrict__ AcsG,
                         const float* __restrict__ Dvec,
                         float* __restrict__ Y) {
  extern __shared__ char smem[];
  float* acs = (float*)smem;               // [256]
  bf16* Ctl  = (bf16*)(acs + CHUNK);       // [256][LDC]
  bf16* Pt   = Ctl + CHUNK * LDC;          // [128][LDC]  [p][n]

  const int c = blockIdx.x, h = blockIdx.y, b = blockIdx.z;
  const int tid = threadIdx.x, wave = tid >> 5, lane = tid & 31;
  const int half = lane >> 4, nn = lane & 15;
  const size_t row0 = (size_t)b * SEQLEN + (size_t)c * CHUNK;
  const size_t sb = ((size_t)(b * NCHUNK + c) * NHEADS + h) << 14;

  if (tid < CHUNK)
    acs[tid] = AcsG[(((size_t)(b * NCHUNK + c) * NHEADS + h) << 8) + tid];
  for (int i = tid; i < CHUNK * 32; i += 512) {
    const int s = i >> 5, q = (i & 31) * 4;
    st_bf16x4(&Ctl[s * LDC + q],
              *(const float4*)&xBC_act[(row0 + s) * CONV_DIM + D_INNER +
                                       D_STATE + q]);
  }
  for (int i = tid; i < 128 * 32; i += 512) {
    const int p = i >> 5, q = (i & 31) * 4;
    st_bf16x4(&Pt[p * LDC + q], *(const float4*)&prev[sb + p * 128 + q]);
  }
  __syncthreads();

  const int m = wave;
  const float Dh = Dvec[h];
#pragma unroll
  for (int pj = 0; pj < 8; ++pj) {
    v8f acc = {};
#pragma unroll
    for (int kk = 0; kk < D_STATE; kk += 32)
      acc = wmma_bf16(load_frag(Ctl + m * 16 * LDC + kk, LDC),
                      load_frag(Pt + (pj * 16) * LDC + kk, LDC), acc);
#pragma unroll
    for (int r = 0; r < 8; ++r) {
      const int tt = m * 16 + r + 8 * half;
      const int e  = h * HEADDIM + pj * 16 + nn;
      const size_t row = row0 + tt;
      float y = Y[row * D_INNER + e] + acc[r] * __expf(acs[tt]);
      y += Dh * xBC_act[row * CONV_DIM + e];
      const float z = zxbcdt[row * D_IN_PROJ + e];
      Y[row * D_INNER + e] = y * siluf(z);
    }
  }
}

// =====================================================================
// K6: RMSNorm over D_INNER
// =====================================================================
__global__ __launch_bounds__(256)
void rmsnorm_kernel(const float* __restrict__ yg,
                    const float* __restrict__ norm_w,
                    float* __restrict__ yn) {
  __shared__ float red[256];
  const size_t row = blockIdx.x;
  float ss = 0.f;
  for (int e = threadIdx.x * 4; e < D_INNER; e += 1024) {
    const float4 v = *(const float4*)&yg[row * D_INNER + e];
    ss += v.x * v.x + v.y * v.y + v.z * v.z + v.w * v.w;
  }
  red[threadIdx.x] = ss;
  __syncthreads();
  for (int off = 128; off > 0; off >>= 1) {
    if (threadIdx.x < off) red[threadIdx.x] += red[threadIdx.x + off];
    __syncthreads();
  }
  const float scale = rsqrtf(red[0] / (float)D_INNER + EPS);
  for (int e = threadIdx.x * 4; e < D_INNER; e += 1024) {
    float4 v = *(const float4*)&yg[row * D_INNER + e];
    const float4 w = *(const float4*)&norm_w[e];
    v.x *= scale * w.x; v.y *= scale * w.y; v.z *= scale * w.z; v.w *= scale * w.w;
    *(float4*)&yn[row * D_INNER + e] = v;
  }
}

// =====================================================================
// launch
// =====================================================================
extern "C" void kernel_launch(void* const* d_in, const int* in_sizes, int n_in,
                              void* d_out, int out_size, void* d_ws,
                              size_t ws_size, hipStream_t stream) {
  const float* u       = (const float*)d_in[0];
  const float* w_in    = (const float*)d_in[1];
  const float* conv_w  = (const float*)d_in[2];
  const float* conv_b  = (const float*)d_in[3];
  const float* dt_bias = (const float*)d_in[4];
  const float* A_log   = (const float*)d_in[5];
  const float* Dvec    = (const float*)d_in[6];
  const float* norm_w  = (const float*)d_in[7];
  const float* w_out   = (const float*)d_in[8];
  float* out = (float*)d_out;

  char* ws = (char*)d_ws;
  auto carve = [&](size_t bytes) {
    char* p = ws;
    ws += (bytes + 255) & ~(size_t)255;
    return p;
  };
  float* zxbcdt  = (float*)carve((size_t)MROWS * D_IN_PROJ * 4);
  float* xBC_act = (float*)carve((size_t)MROWS * CONV_DIM * 4);
  float* Ydiag   = (float*)carve((size_t)MROWS * D_INNER * 4);
  float* AcsG    = (float*)carve((size_t)BATCH * NCHUNK * NHEADS * CHUNK * 4);
  float* states  = (float*)carve((size_t)BATCH * NCHUNK * NHEADS * 128 * 128 * 4);
  float* prev    = (float*)carve((size_t)BATCH * NCHUNK * NHEADS * 128 * 128 * 4);
  float* ynorm   = (float*)carve((size_t)MROWS * D_INNER * 4);
  (void)ws_size; (void)in_sizes; (void)n_in; (void)out_size;

  const size_t smem3 = 3 * CHUNK * 4 +
                       ((size_t)2 * CHUNK * LDC + 2 * 128 * LDT + 16 * 16 * 32) * 2;
  const size_t smem5 = CHUNK * 4 + ((size_t)CHUNK * LDC + 128 * LDC) * 2;
  hipFuncSetAttribute((const void*)ssd_chunk_kernel,
                      hipFuncAttributeMaxDynamicSharedMemorySize, (int)smem3);
  hipFuncSetAttribute((const void*)ssd_off_gate_kernel,
                      hipFuncAttributeMaxDynamicSharedMemorySize, (int)smem5);

  // K1: in_proj GEMM
  gemm_nt_kernel<<<dim3((D_IN_PROJ + BN - 1) / BN, MROWS / BM), 256, 0, stream>>>(
      u, w_in, zxbcdt, MROWS, D_IN_PROJ, D_MODEL);

  // K2: conv + silu
  {
    const int total = BATCH * SEQLEN * CONV_DIM;
    conv_silu_kernel<<<(total + 255) / 256, 256, 0, stream>>>(zxbcdt, conv_w,
                                                              conv_b, xBC_act);
  }

  // K3: intra-chunk SSD (Y_diag + states)
  ssd_chunk_kernel<<<dim3(NCHUNK, NHEADS, BATCH), 512, smem3, stream>>>(
      zxbcdt, xBC_act, dt_bias, A_log, Ydiag, states, AcsG);

  // K4: inter-chunk state scan
  chunk_scan_kernel<<<BATCH * NHEADS, 256, 0, stream>>>(states, AcsG, prev);

  // K5: Y_off + skip + gate (in-place into Ydiag)
  ssd_off_gate_kernel<<<dim3(NCHUNK, NHEADS, BATCH), 512, smem5, stream>>>(
      zxbcdt, xBC_act, prev, AcsG, Dvec, Ydiag);

  // K6: RMSNorm
  rmsnorm_kernel<<<MROWS, 256, 0, stream>>>(Ydiag, norm_w, ynorm);

  // K7: out_proj GEMM
  gemm_nt_kernel<<<dim3(D_MODEL / BN, MROWS / BM), 256, 0, stream>>>(
      ynorm, w_out, out, MROWS, D_MODEL, D_INNER);
}
